// NeighborPruning_69389491634808
// MI455X (gfx1250) — matrix operand order repaired and checked by reading, not compiled, not run on hardware
//
#include <hip/hip_runtime.h>
#include <hip/hip_bf16.h>
#include <climits>

// ---------------------------------------------------------------------------
// NeighborPruning for MI455X (gfx1250, wave32, WMMA + TDM).
//
// LayerNorm over a size-1 axis is degenerate: mu = s exactly, var = 0 exactly,
// so score = 0*rsqrt(eps)*gamma + beta == ln_beta for every edge. All scores
// tie, so the stable segmented top-k reduces to "keep the 3 lowest-indexed
// non-self edges per dst, plus all self-loops" (3 atomicMin passes, exact and
// deterministic). The MLP is still evaluated faithfully with bf16 WMMA — the
// LN collapse makes its precision irrelevant to the output.
//
// Data movement:
//   - h (12.8MB) and q (16KB) are L2-resident (192MB L2); A fragments are
//     gathered directly with b128 loads.
//   - W1 (96KB, pre-swizzled bf16 fragment layout) is staged ONCE per
//     workgroup into LDS by the Tensor Data Mover; each wave grid-strides
//     over 16-edge tiles reading B fragments with ds_load_b128.
//   - Compiler hazards handled explicitly:
//       (a) TDM carries the LDS address as descriptor bits -> sW1 must
//           escape via asm or the ds loads fold to poison (round-3 asm);
//       (b) B-fragment LDS loads are tile-invariant -> LICM hoists 96
//           fragments and spills to scratch (round-4 asm); an opaque
//           per-iteration LDS offset defeats the hoist;
//       (c) ds_load immediate offsets are 16-bit; an OPAQUE +64KB offset
//           (not CSE-able back to base+imm, round-6 asm) gives a second base
//           register so high fragments avoid in-chain v_add + WAR-nop bursts;
//       (d) one-deep B-fragment software pipeline keeps two ds loads in
//           flight behind each WMMA (s_wait_dscnt 2, not 0).
// ---------------------------------------------------------------------------

typedef __attribute__((ext_vector_type(16))) __bf16 v16bf;
typedef __attribute__((ext_vector_type(8)))  float  v8f;
typedef __attribute__((ext_vector_type(4)))  unsigned int u32x4;
typedef __attribute__((ext_vector_type(8)))  int i32x8;
typedef __attribute__((ext_vector_type(4)))  int i32x4;

union BFrag { uint4 u[2]; v16bf v; };

#define W1_ELEMS  (192 * 256)          // 49152 bf16 = 96KB
#define SCORE_BLOCKS 1024              // fixed grid: grid-stride + LDS amortization

__device__ __forceinline__ unsigned short f2bf_bits(float f) {
  __bf16 hb = (__bf16)f;               // RNE
  unsigned short u;
  __builtin_memcpy(&u, &hb, sizeof(u));
  return u;
}

// ---- workspace init: per-dst 3 smallest non-self edge indices --------------
__global__ void np_init_mins(int* __restrict__ mins, int n) {
  int i = blockIdx.x * blockDim.x + threadIdx.x;
  if (i < n) mins[i] = INT_MAX;
}

// ---- pre-swizzle W1 (192x256 f32 row-major) into WMMA B-fragment layout ----
__global__ void np_w1_swizzle(const float* __restrict__ W1,
                              unsigned short* __restrict__ W1swz) {
  int t = blockIdx.x * blockDim.x + threadIdx.x;
  if (t >= W1_ELEMS) return;
  int k = t >> 8, n = t & 255;
  unsigned short b = f2bf_bits(W1[t]);
  int kstep = k >> 5, kin = k & 31;
  int hi = kin >> 4, kk = kin & 15;
  int vreg = kk >> 1, odd = kk & 1;
  int ntile = n >> 4, nlo = n & 15;
  int lane = hi * 16 + nlo;
  int frag = kstep * 16 + ntile;
  W1swz[frag * 512 + lane * 16 + vreg * 2 + odd] = b;
}

// ---- pass p: smallest non-self edge index per dst, excluding prior minima --
__global__ void np_min_pass(const int* __restrict__ src, const int* __restrict__ dst,
                            int* __restrict__ mins, int N, int E, int pass) {
  int e = blockIdx.x * blockDim.x + threadIdx.x;
  if (e >= E) return;
  int s = src[e], d = dst[e];
  if (s == d) return;
  if (pass > 0 && mins[d] == e) return;
  if (pass > 1 && mins[N + d] == e) return;
  atomicMin(&mins[pass * N + d], e);
}

__global__ void np_finalize_keep(const int* __restrict__ src, const int* __restrict__ dst,
                                 const int* __restrict__ mins, int N, int E,
                                 float* __restrict__ keep) {
  int e = blockIdx.x * blockDim.x + threadIdx.x;
  if (e >= E) return;
  int s = src[e], d = dst[e];
  bool k = (s == d) || (mins[d] == e) || (mins[N + d] == e) || (mins[2 * N + d] == e);
  keep[e] = k ? 1.0f : 0.0f;
}

// ---- faithful MLP scores: TDM-staged W1 in LDS, bf16 WMMA, literal LN ------
__global__ void np_score_wmma(const float* __restrict__ h, const float* __restrict__ q,
                              const int* __restrict__ src, const int* __restrict__ dst,
                              const int* __restrict__ eb,
                              const unsigned short* __restrict__ W1swz,
                              const float* __restrict__ b1, const float* __restrict__ W2,
                              const float* __restrict__ b2, const float* __restrict__ ln_g,
                              const float* __restrict__ ln_b,
                              float* __restrict__ scores, int E) {
  __shared__ unsigned short sW1[W1_ELEMS];   // 96KB (<= 320KB/WG on CDNA5)

  int lane = threadIdx.x & 31;
  int wave = threadIdx.x >> 5;

  // --- Stage W1swz (96KB) into LDS with the Tensor Data Mover, wave 0 only.
  // 2D D#: 4096 dwords x 6 rows, data_size=4B, lds_addr=0 (sole shared array),
  // global_addr=W1swz, type=2. Tensor spans exactly our valid workspace.
  if (wave == 0) {
    unsigned long long ga = (unsigned long long)(uintptr_t)W1swz;
    u32x4 g0;
    g0[0] = 1u;                                            // count=1, user mode
    g0[1] = 0u;                                            // lds_addr = 0
    g0[2] = (unsigned)(ga & 0xFFFFFFFFull);                // global_addr[31:0]
    g0[3] = (unsigned)((ga >> 32) & 0x1FFFFFFull) | (2u << 30);  // addr[56:32] | type=2
    i32x8 g1;
    g1[0] = (int)(2u << 16);          // wgmask=0, data_size=2 (4B), no flags
    g1[1] = (int)(4096u << 16);       // atomic_barrier_addr=0 | tensor_dim0[15:0]=4096
    g1[2] = (int)(6u << 16);          // tensor_dim0[31:16]=0 | tensor_dim1[15:0]=6
    g1[3] = (int)(4096u << 16);       // tensor_dim1[31:16]=0 | tile_dim0=4096
    g1[4] = 6;                        // tile_dim1=6, tile_dim2=0
    g1[5] = 4096;                     // tensor_dim0_stride[31:0]=4096
    g1[6] = 0;                        // stride0[47:32]=0 | stride1[15:0]=0
    g1[7] = 0;
    i32x4 gz4 = {0, 0, 0, 0};
    i32x8 gz8 = {0, 0, 0, 0, 0, 0, 0, 0};
    __builtin_amdgcn_tensor_load_to_lds(g0, g1, gz4, gz4, gz8, 0);
    __builtin_amdgcn_s_wait_tensorcnt(0);
  }

  int nlo = lane & 15;               // A-row M / C-column N for this lane
  int hi = lane >> 4;                // K-half selector

  // Tile-invariant epilogue weights, preloaded while the TDM streams W1.
  float b1v[16], w2v[16];
#pragma unroll
  for (int n2 = 0; n2 < 16; ++n2) {
    b1v[n2] = b1[(n2 << 4) + nlo];
    w2v[n2] = W2[(n2 << 4) + nlo];
  }
  float b2v = b2[0], gv = ln_g[0], bv = ln_b[0];

  // TDM wrote sW1 behind the compiler's back; force the array to escape.
  asm volatile("" : : "r"(&sW1[0]) : "memory");
  __syncthreads();

  int nt = (E + 15) >> 4;

  for (int tile = blockIdx.x * 8 + wave; tile < nt; tile += gridDim.x * 8) {
    // Opaque zero, redefined every iteration: defeats LICM of the B loads
    // (round-4 asm showed 96 hoisted fragments spilling to scratch).
    unsigned lds_off = 0;
    asm volatile("" : "+v"(lds_off));
    // Opaque 64KB: base_hi cannot be rematerialized as base_lo + imm inside
    // the WMMA chain (round-6 asm showed v_add + 4x v_nop WAR bursts there).
    unsigned hi_off = 65536;
    asm volatile("" : "+v"(hi_off));

    // Dual LDS bases keep every ds_load offset < 64KB (16-bit immediate).
    // Only integer offsets are laundered, so the pointer chain still roots at
    // the __shared__ array and the loads stay ds_load (not flat).
    const char* base_lo = (const char*)sW1 + (unsigned)(lane * 32) + lds_off;
    const char* base_hi = base_lo + hi_off;

    int e0 = tile << 4;
    int eM = e0 + nlo; if (eM > E - 1) eM = E - 1;   // tail clamp; stores guarded
    int sM = src[eM], dM = dst[eM], bM = eb[eM];
    const float* rp0 = h + (size_t)sM * 64;
    const float* rp1 = h + (size_t)dM * 64;
    const float* rp2 = q + (size_t)bM * 64;

    // A fragments for all 6 k-steps (K = 192).
    v16bf A[6];
#pragma unroll
    for (int ks = 0; ks < 6; ++ks) {
#pragma unroll
      for (int run = 0; run < 2; ++run) {
        int f = ks * 32 + hi * 16 + run * 8;        // 8 consecutive features
        const float* p = (f < 64) ? (rp0 + f)
                       : (f < 128) ? (rp1 + (f - 64))
                                   : (rp2 + (f - 128));
        float4 a = *(const float4*)(p);
        float4 c = *(const float4*)(p + 4);
        int base = run * 8;
        A[ks][base + 0] = (__bf16)a.x; A[ks][base + 1] = (__bf16)a.y;
        A[ks][base + 2] = (__bf16)a.z; A[ks][base + 3] = (__bf16)a.w;
        A[ks][base + 4] = (__bf16)c.x; A[ks][base + 5] = (__bf16)c.y;
        A[ks][base + 6] = (__bf16)c.z; A[ks][base + 7] = (__bf16)c.w;
      }
    }

    float ps[8];
#pragma unroll
    for (int r = 0; r < 8; ++r) ps[r] = 0.0f;

    // hid = relu(x @ W1 + b1); 256->1 projection folded into the epilogue.
    // One-deep B pipeline: next fragment's ds loads issue before this WMMA.
    // All offsets are compile-time constants after full unroll.
#define LOAD_B(B_, fo_)                                                        \
    do {                                                                       \
      const char* bp_ = ((fo_) >= 65536) ? base_hi : base_lo;                  \
      int o_ = ((fo_) >= 65536) ? ((fo_) - 65536) : (fo_);                     \
      (B_).u[0] = *(const uint4*)(bp_ + o_);                                   \
      (B_).u[1] = *(const uint4*)(bp_ + o_ + 16);                              \
    } while (0)

    BFrag Bc;
    LOAD_B(Bc, 0);                     // frag (ks=0, ntile=0)
#pragma unroll
    for (int ntile = 0; ntile < 16; ++ntile) {
      v8f acc = {};
#pragma unroll
      for (int ks = 0; ks < 6; ++ks) {
        BFrag Bn = Bc;
        if (!(ntile == 15 && ks == 5)) {
          int nks = (ks == 5) ? 0 : ks + 1;
          int nnt = (ks == 5) ? ntile + 1 : ntile;
          LOAD_B(Bn, (nks * 16 + nnt) * 1024);   // next fragment in flight
        }
        acc = __builtin_amdgcn_wmma_f32_16x16x32_bf16(
            false, A[ks], false, Bc.v, (short)0, acc, false, false);
        Bc = Bn;
      }
      // C layout: VGPR r, lane L -> (M = r + 8*hi, N = L&15)
      float b1n = b1v[ntile];
      float w2n = w2v[ntile];
#pragma unroll
      for (int r = 0; r < 8; ++r) {
        float v = acc[r] + b1n;
        v = v > 0.0f ? v : 0.0f;
        ps[r] = fmaf(v, w2n, ps[r]);
      }
    }
#undef LOAD_B

    // Reduce over N: xor masks 1,2,4,8 stay within each 16-lane half.
#pragma unroll
    for (int m = 1; m < 16; m <<= 1) {
#pragma unroll
      for (int r = 0; r < 8; ++r) ps[r] += __shfl_xor(ps[r], m, 32);
    }

    if (nlo == 0) {
#pragma unroll
      for (int r = 0; r < 8; ++r) {
        int e = e0 + hi * 8 + r;
        if (e < E) {
          float s = ps[r] + b2v;
          // Literal LayerNorm over a size-1 axis (mu == s; not foldable w/o fast-math).
          float mu  = s;
          float dd  = s - mu;       // exactly 0 for finite s
          float var = dd * dd;      // exactly 0
          scores[e] = dd * rsqrtf(var + 1e-5f) * gv + bv;   // == ln_beta
        }
      }
    }
  }
}

// ---------------------------------------------------------------------------
extern "C" void kernel_launch(void* const* d_in, const int* in_sizes, int n_in,
                              void* d_out, int out_size, void* d_ws, size_t ws_size,
                              hipStream_t stream) {
  const float* h   = (const float*)d_in[0];
  const float* q   = (const float*)d_in[1];
  const int*   ei  = (const int*)d_in[2];   // int32 (JAX default canonicalizes i64)
  const int*   eb  = (const int*)d_in[3];
  const float* W1  = (const float*)d_in[4];
  const float* b1  = (const float*)d_in[5];
  const float* W2  = (const float*)d_in[6];
  const float* b2  = (const float*)d_in[7];
  const float* lng = (const float*)d_in[8];
  const float* lnb = (const float*)d_in[9];

  int N = in_sizes[0] / 64;                 // 50000
  int E = in_sizes[2] / 2;                  // 400000
  const int* src = ei;
  const int* dst = ei + E;

  float* keep   = (float*)d_out;            // [E] keep, then [E] scores
  float* scores = keep + E;

  // Workspace: 3*N ints (per-dst 3 smallest indices) + swizzled bf16 W1.
  int* mins = (int*)d_ws;
  size_t off = (((size_t)3 * N * sizeof(int)) + 255) & ~(size_t)255;
  unsigned short* W1swz = (unsigned short*)((char*)d_ws + off);

  np_init_mins<<<(3 * N + 255) / 256, 256, 0, stream>>>(mins, 3 * N);
  np_w1_swizzle<<<(W1_ELEMS + 255) / 256, 256, 0, stream>>>(W1, W1swz);

  for (int pass = 0; pass < 3; ++pass) {
    np_min_pass<<<(E + 255) / 256, 256, 0, stream>>>(src, dst, mins, N, E, pass);
  }

  int nt = (E + 15) >> 4;                   // 16-edge tiles
  int blocks = (nt + 7) / 8;
  if (blocks > SCORE_BLOCKS) blocks = SCORE_BLOCKS;
  if (blocks < 1) blocks = 1;
  np_score_wmma<<<blocks, 256, 0, stream>>>(h, q, src, dst, eb, W1swz,
                                            b1, W2, b2, lng, lnb, scores, E);

  np_finalize_keep<<<(E + 255) / 256, 256, 0, stream>>>(src, dst, mins, N, E, keep);
}
